// GTTFNLayer_36043365548221
// MI455X (gfx1250) — matrix operand order repaired and checked by reading, not compiled, not run on hardware
//
#include <hip/hip_runtime.h>

typedef __attribute__((ext_vector_type(16))) _Float16 v16h;
typedef __attribute__((ext_vector_type(8)))  float    v8f;

#define N_    512
#define C_    8
#define NRBF  32
#define NINT  15

struct AInter { const float* feat; const float* cg; int di, de, dd, fcgoff; };
struct AParams { AInter it[NINT]; };

struct BInter { const float *w1, *b1, *w2, *b2; int fcgoff, moff, eoff; };
struct BParams { BInter it[NINT]; };

struct NParams { int cnt[3]; int offs[3][6]; };

// ---------------- Kernel 1: fCG[j,c,e,d] = sum_in feat_li[j,c,in] * cg[in,e,d] ----
__global__ void fcg_kernel(AParams P, float* __restrict__ fcg) {
  const AInter a = P.it[blockIdx.y];
  const int count = N_ * C_ * a.de * a.dd;
  const int idx = blockIdx.x * blockDim.x + threadIdx.x;
  if (idx >= count) return;
  const int d  = idx % a.dd;
  const int e  = (idx / a.dd) % a.de;
  const int jc = idx / (a.dd * a.de);              // j*8 + c
  float acc = 0.f;
  for (int in = 0; in < a.di; ++in)
    acc += a.feat[jc * a.di + in] * a.cg[(in * a.de + e) * a.dd + d];
  fcg[a.fcgoff + idx] = acc;
}

// ---------------- Kernel 2: per-(i, interaction) fused MLP + contraction --------
// Templated on DE (edge dim) / DD (output dim) so every stride and trip count
// constant-folds; 9 instantiations cover the 15 interactions.
template<int DE, int DD>
__global__ __launch_bounds__(32)
void main_kernel(BParams P, const float* __restrict__ rbf,
                 const float* __restrict__ gt_edge,
                 const unsigned char* __restrict__ mask,
                 const float* __restrict__ fcg,
                 float* __restrict__ msgws) {
  const int i = blockIdx.x;
  const BInter ip = P.it[blockIdx.y];
  const int L   = threadIdx.x;
  const int l16 = L & 15;
  const bool hi = (L >= 16);

  __shared__ _Float16 Hs[16 * 32];    // h tile, row-major [j][hid]
  __shared__ _Float16 RT[64 * 16];    // radial transposed [col=c*8+o][j]
  __shared__ _Float16 MT[8 * 128];    // masked transposed [d][k=c*16+j]
  __shared__ float    Es[16 * 6];     // per-j: e_feat[0..DE-1], mask at [5]

  // ---- preload W1 (2 B-tiles), W2 (4 B-tiles), biases; B layout: col=lane%16,
  // lanes 0-15 hold K=0..15 (2 f16/VGPR), lanes 16-31 hold K=16..31.
  const int kb16 = hi ? 16 : 0;
  v16h bw1[2]; v16h bw2[4];
  float b1v[2], b2v[4];
  for (int tt = 0; tt < 2; ++tt) {
    for (int q = 0; q < 16; ++q)
      bw1[tt][q] = (_Float16)ip.w1[(kb16 + q) * 32 + tt * 16 + l16];
    b1v[tt] = ip.b1[tt * 16 + l16];
  }
  for (int tt = 0; tt < 4; ++tt) {
    for (int q = 0; q < 16; ++q)
      bw2[tt][q] = (_Float16)ip.w2[(kb16 + q) * 64 + tt * 16 + l16];
    b2v[tt] = ip.b2[tt * 16 + l16];
  }

  v8f macc = {0.f, 0.f, 0.f, 0.f, 0.f, 0.f, 0.f, 0.f};
  const int rowD = hi ? 8 : 0;   // D-matrix row offset for this half-wave
  const int klo  = hi ? 16 : 0;  // A-operand K base for this half-wave

  // masked-loop lane decomposition: entry = L*32 + t, t<32 never crosses a
  // 128 boundary -> d is constant per lane, c takes two values.
  const int dlane = L >> 2;
  const int cbase = (L & 3) * 2;

  for (int j0 = 0; j0 < N_; j0 += 16) {
    // ---- stage e_feat + mask for this tile (lanes 0-15, one j each) ----
    if (L < 16) {
      const size_t ej = ((size_t)i * N_ + (j0 + L)) * 9 + ip.eoff;
      #pragma unroll
      for (int e = 0; e < DE; ++e) Es[L * 6 + e] = gt_edge[ej + e];
      Es[L * 6 + 5] = (float)mask[(size_t)i * N_ + j0 + L];
    }

    // ---- A tile: rbf[i, j0+jl, 0..31] contiguous K: element s holds k = klo + s ----
    const float* rrow = rbf + ((size_t)i * N_ + (j0 + l16)) * NRBF;
    if (j0 + 16 < N_) __builtin_prefetch(rrow + 16 * NRBF, 0, 3);
    v16h a;
    #pragma unroll
    for (int q = 0; q < 8; ++q) {
      a[q]     = (_Float16)rrow[klo + q];
      a[8 + q] = (_Float16)rrow[klo + 8 + q];
    }

    // ---- h = silu(A @ W1 + b1), fp32 accumulate ----
    v8f h0 = {}, h1 = {};
    h0 = __builtin_amdgcn_wmma_f32_16x16x32_f16(false, a, false, bw1[0], (short)0, h0, false, false);
    h1 = __builtin_amdgcn_wmma_f32_16x16x32_f16(false, a, false, bw1[1], (short)0, h1, false, false);
    #pragma unroll
    for (int r = 0; r < 8; ++r) {
      float x0 = h0[r] + b1v[0];
      float x1 = h1[r] + b1v[1];
      x0 = x0 / (1.f + __expf(-x0));
      x1 = x1 / (1.f + __expf(-x1));
      const int row = r + rowD;
      Hs[row * 32 + l16]      = (_Float16)x0;
      Hs[row * 32 + 16 + l16] = (_Float16)x1;
    }
    __syncthreads();

    // ---- rebuild A' from Hs; radial = A' @ W2 + b2 -> RT (transposed f16) ----
    v16h a2;
    #pragma unroll
    for (int q = 0; q < 8; ++q) {
      a2[q]     = Hs[l16 * 32 + klo + q];
      a2[8 + q] = Hs[l16 * 32 + klo + 8 + q];
    }
    #pragma unroll
    for (int tt = 0; tt < 4; ++tt) {
      v8f rv = {};
      rv = __builtin_amdgcn_wmma_f32_16x16x32_f16(false, a2, false, bw2[tt], (short)0, rv, false, false);
      const int col = tt * 16 + l16;
      #pragma unroll
      for (int r = 0; r < 8; ++r)
        RT[col * 16 + r + rowD] = (_Float16)(rv[r] + b2v[tt]);
    }

    // ---- masked[j,c,d] = m * sum_e e_feat[e]*fCG[j,c,e,d] -> MT[d][c*16+jl] ----
    if (dlane < DD) {
      #pragma unroll
      for (int half = 0; half < 2; ++half) {
        const int c = cbase + half;
        const float* fbase = fcg + ip.fcgoff + dlane +
            (size_t)(j0 * 8 + c) * (DE * DD);
        for (int jl = 0; jl < 16; ++jl) {
          const float mv = Es[jl * 6 + 5];
          float val = 0.f;
          if (mv != 0.f) {
            const float* fp = fbase + (size_t)jl * (8 * DE * DD);
            float acc = 0.f;
            #pragma unroll
            for (int e = 0; e < DE; ++e) acc += Es[jl * 6 + e] * fp[e * DD];
            val = mv * acc;
          }
          MT[dlane * 128 + c * 16 + jl] = (_Float16)val;
        }
      }
    } else {
      #pragma unroll
      for (int half = 0; half < 2; ++half)
        for (int jl = 0; jl < 16; ++jl)
          MT[dlane * 128 + (cbase + half) * 16 + jl] = (_Float16)0.f;
    }
    __syncthreads();

    // ---- msg[o,d] += radial^T @ masked, K = 128 per tile (4 WMMAs) ----
    const int oc = l16 & 7;                    // clamp: rows/cols >=8 are ignored
    const int jb = hi ? 8 : 0;
    #pragma unroll
    for (int kk = 0; kk < 4; ++kk) {
      v16h ak, bk;
      const int c0 = 2 * kk, c1 = 2 * kk + 1;
      #pragma unroll
      for (int q = 0; q < 8; ++q) {
        ak[q]     = RT[(c0 * 8 + oc) * 16 + jb + q];
        ak[8 + q] = RT[(c1 * 8 + oc) * 16 + jb + q];
      }
      const int kbB = kk * 32 + (hi ? 8 : 0);
      #pragma unroll
      for (int q = 0; q < 16; ++q) bk[q] = MT[oc * 128 + kbB + q + (q >= 8 ? 8 : 0)];
      macc = __builtin_amdgcn_wmma_f32_16x16x32_f16(false, ak, false, bk, (short)0, macc, false, false);
    }
    __syncthreads();   // protect LDS before next tile overwrites
  }

  // ---- write partial msg (exclusive slot: no atomics, deterministic) ----
  if (!hi && l16 < DD) {
    float* ob = msgws + ip.moff + (size_t)i * C_ * DD + l16;
    #pragma unroll
    for (int r = 0; r < 8; ++r) ob[r * DD] = macc[r];
  }
}

// ---------------- Kernel 3: sum partials per lo + channel layer-norm ------------
__global__ void norm_kernel(NParams NP, const float* __restrict__ msgws,
                            float* __restrict__ out,
                            const float* g0, const float* be0,
                            const float* g1, const float* be1,
                            const float* g2, const float* be2) {
  const int idx = blockIdx.x * blockDim.x + threadIdx.x;
  if (idx >= N_ * 3) return;
  const int i = idx / 3, lo = idx % 3;
  const int dd    = (lo == 0) ? 1 : (lo == 1) ? 3 : 5;
  const int obase = (lo == 0) ? 0 : (lo == 1) ? 4096 : 16384;
  const float* gg = (lo == 0) ? g0 : (lo == 1) ? g1 : g2;
  const float* bb = (lo == 0) ? be0 : (lo == 1) ? be1 : be2;

  float f[8][5], nrm[8];
  float mu = 0.f;
  for (int c = 0; c < 8; ++c) {
    float s = 0.f;
    for (int d = 0; d < dd; ++d) {
      float v = 0.f;
      for (int s2 = 0; s2 < NP.cnt[lo]; ++s2)
        v += msgws[NP.offs[lo][s2] + ((size_t)i * 8 + c) * dd + d];
      f[c][d] = v;
      s += v * v;
    }
    nrm[c] = fmaxf(sqrtf(s), 1e-8f);
    mu += nrm[c];
  }
  mu *= 0.125f;
  float var = 0.f;
  for (int c = 0; c < 8; ++c) { float t = nrm[c] - mu; var += t * t; }
  var *= 0.125f;
  const float inv = rsqrtf(var + 1e-5f);
  float* base = out + obase + (size_t)i * 8 * dd;
  for (int c = 0; c < 8; ++c) {
    const float sc = (nrm[c] - mu) * inv * gg[c] + bb[c];
    for (int d = 0; d < dd; ++d) base[c * dd + d] = f[c][d] / nrm[c] * sc;
  }
}

extern "C" void kernel_launch(void* const* d_in, const int* in_sizes, int n_in,
                              void* d_out, int out_size, void* d_ws, size_t ws_size,
                              hipStream_t stream) {
  static const int LI[NINT] = {0,0,0,1,1,1,1,1,1,2,2,2,2,2,2};
  static const int LE[NINT] = {0,1,2,0,1,1,1,2,2,0,1,1,2,2,2};
  static const int LO[NINT] = {0,1,2,1,0,1,2,1,2,2,1,2,0,1,2};
  static const int DIMS[3]  = {1, 3, 5};
  static const int EOFF[3]  = {0, 1, 4};

  AParams ap; BParams bp; NParams np;
  np.cnt[0] = np.cnt[1] = np.cnt[2] = 0;

  int fcg_off = 0;
  for (int t = 0; t < NINT; ++t)
    fcg_off += N_ * C_ * DIMS[LE[t]] * DIMS[LO[t]];
  const int fcg_total = fcg_off;        // floats
  int moff = fcg_total;

  fcg_off = 0;
  for (int t = 0; t < NINT; ++t) {
    const int li = LI[t], le = LE[t], lo = LO[t];
    const int di = DIMS[li], de = DIMS[le], dd = DIMS[lo];
    const int pb = 6 + 5 * t;   // cg, w1, b1, w2, b2
    ap.it[t].feat = (const float*)d_in[li];
    ap.it[t].cg   = (const float*)d_in[pb];
    ap.it[t].di = di; ap.it[t].de = de; ap.it[t].dd = dd; ap.it[t].fcgoff = fcg_off;

    bp.it[t].w1 = (const float*)d_in[pb + 1];
    bp.it[t].b1 = (const float*)d_in[pb + 2];
    bp.it[t].w2 = (const float*)d_in[pb + 3];
    bp.it[t].b2 = (const float*)d_in[pb + 4];
    bp.it[t].fcgoff = fcg_off;
    bp.it[t].moff = moff; bp.it[t].eoff = EOFF[le];

    np.offs[lo][np.cnt[lo]++] = moff;

    fcg_off += N_ * C_ * de * dd;
    moff    += N_ * C_ * dd;
  }
  (void)in_sizes; (void)n_in; (void)out_size; (void)ws_size;

  float* fcg   = (float*)d_ws;
  float* msgws = (float*)d_ws;          // msg slots start at fcg_total (offsets absolute)
  float* outp  = (float*)d_out;
  const float* rbfp = (const float*)d_in[3];
  const float* gtep = (const float*)d_in[4];
  const unsigned char* maskp = (const unsigned char*)d_in[5];

  // K1: fCG (max per-interaction count = 512*8*25 = 102400 -> 400 blocks of 256)
  fcg_kernel<<<dim3(400, NINT), 256, 0, stream>>>(ap, fcg);

  // K2: group interactions by (de,dd) -> 9 specialized launches
  int gde[9], gdd[9], gcnt[9], gts[9][3], ng = 0;
  for (int t = 0; t < NINT; ++t) {
    const int de = DIMS[LE[t]], dd = DIMS[LO[t]];
    int g = -1;
    for (int k = 0; k < ng; ++k) if (gde[k] == de && gdd[k] == dd) g = k;
    if (g < 0) { g = ng++; gde[g] = de; gdd[g] = dd; gcnt[g] = 0; }
    gts[g][gcnt[g]++] = t;
  }
  for (int g = 0; g < ng; ++g) {
    BParams gp = bp;
    for (int k = 0; k < gcnt[g]; ++k) gp.it[k] = bp.it[gts[g][k]];
    dim3 grid(N_, gcnt[g]);
    const int de = gde[g], dd = gdd[g];
    #define LAUNCH_MAIN(DE, DD) \
      main_kernel<DE, DD><<<grid, 32, 0, stream>>>(gp, rbfp, gtep, maskp, fcg, msgws)
    if      (de == 1 && dd == 1) LAUNCH_MAIN(1, 1);
    else if (de == 1 && dd == 3) LAUNCH_MAIN(1, 3);
    else if (de == 1 && dd == 5) LAUNCH_MAIN(1, 5);
    else if (de == 3 && dd == 1) LAUNCH_MAIN(3, 1);
    else if (de == 3 && dd == 3) LAUNCH_MAIN(3, 3);
    else if (de == 3 && dd == 5) LAUNCH_MAIN(3, 5);
    else if (de == 5 && dd == 1) LAUNCH_MAIN(5, 1);
    else if (de == 5 && dd == 3) LAUNCH_MAIN(5, 3);
    else if (de == 5 && dd == 5) LAUNCH_MAIN(5, 5);
    #undef LAUNCH_MAIN
  }

  // K3: reduce partials + layer norm -> d_out
  norm_kernel<<<6, 256, 0, stream>>>(np, msgws, outp,
      (const float*)d_in[81], (const float*)d_in[82],
      (const float*)d_in[83], (const float*)d_in[84],
      (const float*)d_in[85], (const float*)d_in[86]);
}
